// ExpertLayer_45122926412361
// MI455X (gfx1250) — compile-verified, matrix operand order checked by reading, not run on hardware
//
#include <hip/hip_runtime.h>
#include <hip/hip_bf16.h>

#define D_MODEL   1024
#define NUM_E     8
#define F_DIM     2048
#define TWO_F     4096
#define N_TOK     8192
#define LN_EPS    1e-5f
#define BAL_COEF  0.01f

typedef __attribute__((ext_vector_type(16))) __bf16 v16bf;
typedef __attribute__((ext_vector_type(8)))  float  v8f;
typedef __attribute__((ext_vector_type(4)))  unsigned int u32x4;

union Frag { v16bf v; u32x4 q[2]; };

__device__ __forceinline__ v8f wmma_bf16(v16bf a, v16bf b, v8f c) {
  return __builtin_amdgcn_wmma_f32_16x16x32_bf16(false, a, false, b, (short)0, c, false, false);
}

// ---------------- one-time weight conversion (f32 -> bf16, transposed) ----------------
// W1[e][d][f] -> W1t[e][f][d]   (B columns contiguous along K=d)
__global__ void __launch_bounds__(256)
k_tw1(const float* __restrict__ w1, __bf16* __restrict__ w1t) {
  __shared__ float tile[32][33];
  const int e  = blockIdx.z;
  const int f0 = blockIdx.x * 32;
  const int d0 = blockIdx.y * 32;
  const int tx = threadIdx.x & 31, ty = threadIdx.x >> 5;
#pragma unroll
  for (int i = 0; i < 4; ++i) {
    int d = d0 + ty + i * 8;
    tile[ty + i * 8][tx] = w1[((size_t)e * D_MODEL + d) * TWO_F + f0 + tx];
  }
  __syncthreads();
#pragma unroll
  for (int i = 0; i < 4; ++i) {
    int f = f0 + ty + i * 8;
    w1t[((size_t)e * TWO_F + f) * D_MODEL + d0 + tx] = (__bf16)tile[tx][ty + i * 8];
  }
}

// Wp[f][d] -> Wpt[d][f]
__global__ void __launch_bounds__(256)
k_twp(const float* __restrict__ wp, __bf16* __restrict__ wpt) {
  __shared__ float tile[32][33];
  const int d0 = blockIdx.x * 32;
  const int f0 = blockIdx.y * 32;
  const int tx = threadIdx.x & 31, ty = threadIdx.x >> 5;
#pragma unroll
  for (int i = 0; i < 4; ++i)
    tile[ty + i * 8][tx] = wp[((size_t)(f0 + ty + i * 8)) * D_MODEL + d0 + tx];
  __syncthreads();
#pragma unroll
  for (int i = 0; i < 4; ++i)
    wpt[((size_t)(d0 + ty + i * 8)) * F_DIM + f0 + tx] = (__bf16)tile[tx][ty + i * 8];
}

__global__ void k_cx(const float* __restrict__ x, __bf16* __restrict__ xbf, int n) {
  int i = blockIdx.x * blockDim.x + threadIdx.x;
  if (i < n) xbf[i] = (__bf16)x[i];
}

// ---------------- routing ----------------
__global__ void k_zero(int* p) { if (threadIdx.x < 32) p[threadIdx.x] = 0; }

// one wave (32 lanes) per token; exact f32 gating so argmax matches reference
__global__ void __launch_bounds__(256)
k_gate(const float* __restrict__ x, const float* __restrict__ wg1,
       const float* __restrict__ bg1, const float* __restrict__ wg2,
       int* __restrict__ choice, int* __restrict__ counts) {
  const int wave = threadIdx.x >> 5, lane = threadIdx.x & 31;
  const int n = blockIdx.x * 8 + wave;
  float acc[16];
#pragma unroll
  for (int j = 0; j < 16; ++j) acc[j] = 0.f;
  const float* xr = x + (size_t)n * D_MODEL;
  for (int d = lane; d < D_MODEL; d += 32) {
    const float xv = xr[d];
    const float4* w = (const float4*)(wg1 + d * 16);
#pragma unroll
    for (int j4 = 0; j4 < 4; ++j4) {
      float4 wv = w[j4];
      acc[j4 * 4 + 0] += xv * wv.x;
      acc[j4 * 4 + 1] += xv * wv.y;
      acc[j4 * 4 + 2] += xv * wv.z;
      acc[j4 * 4 + 3] += xv * wv.w;
    }
  }
#pragma unroll
  for (int off = 16; off >= 1; off >>= 1)
#pragma unroll
    for (int j = 0; j < 16; ++j) acc[j] += __shfl_xor(acc[j], off, 32);
  if (lane == 0) {
    float g[8];
#pragma unroll
    for (int j = 0; j < 8; ++j) {
      float a = acc[j] + bg1[j];
      float b = acc[j + 8] + bg1[j + 8];
      g[j] = a / (1.f + __expf(-b));      // GLU
    }
    int best = 0; float bl = -3.4e38f;
    for (int k = 0; k < 8; ++k) {          // softmax is monotone: argmax(logits)
      float l = 0.f;
#pragma unroll
      for (int j = 0; j < 8; ++j) l += g[j] * wg2[j * 8 + k];
      if (l > bl) { bl = l; best = k; }
    }
    choice[n] = best;
    atomicAdd(counts + best, 1);
  }
}

__global__ void k_loss(const int* __restrict__ counts, float* __restrict__ out_loss) {
  if (threadIdx.x == 0) {
    float loss = 0.f;
    for (int e = 0; e < NUM_E; ++e) {
      float p = (float)counts[e] / (float)N_TOK;
      loss -= p * logf(p + 1e-10f);
    }
    *out_loss = loss * BAL_COEF;
  }
}

__global__ void k_sort(const int* __restrict__ choice, int* __restrict__ cursors,
                       int* __restrict__ sorted) {
  int n = blockIdx.x * blockDim.x + threadIdx.x;
  if (n >= N_TOK) return;
  int e = choice[n];
  int pos = atomicAdd(cursors + e, 1);
  sorted[e * N_TOK + pos] = n;
}

__global__ void k_pad(const int* __restrict__ counts, int* __restrict__ padded,
                      int* __restrict__ sorted) {
  int e = threadIdx.x;
  if (e >= NUM_E) return;
  int c = counts[e];
  int pc = (c + 31) & ~31;               // pad to 32-token tiles (region is 8192, safe)
  padded[e] = pc;
  for (int p = c; p < pc; ++p) sorted[e * N_TOK + p] = -1;
}

// ---------------- routed expert FFN: GEMM1 + GLU + LN stats (bf16 WMMA) ----------------
// block = 32 tokens of one expert; 8 waves; per wave: 2 row-subtiles x (a-col16, gate-col16)
// -> 4 WMMA accumulators, B fragments reused across both row-subtiles (halves B traffic).
// GLU done in registers; un-normalized bf16 written to hbf; LN mean/var accumulated in
// registers per lane, reduced via shfl_xor + tiny LDS atomics; k_norm applies LN after.
__global__ void __launch_bounds__(256)
k_ffn(const __bf16* __restrict__ xbf, const __bf16* __restrict__ w1t,
      const float* __restrict__ b1, const int* __restrict__ sorted,
      const int* __restrict__ padded, __bf16* __restrict__ hbf,
      float* __restrict__ stats) {
  __shared__ int   toks[32];
  __shared__ float sums[32], sumsq[32];

  const int e  = blockIdx.y;
  const int rt = blockIdx.x;
  if (rt * 32 >= padded[e]) return;      // uniform early-exit for inactive tiles

  const int tid = threadIdx.x, wave = tid >> 5, lane = tid & 31;
  if (tid < 32) {
    toks[tid] = sorted[e * N_TOK + rt * 32 + tid];
    sums[tid] = 0.f; sumsq[tid] = 0.f;
  }
  __syncthreads();

  const int m16 = lane & 15;
  const int hi  = lane >> 4;
  const int kb  = hi * 8;                // per-lane K sub-offset (ISA 16-bit A/B layout)
  const int t0  = toks[m16];
  const int t1  = toks[m16 + 16];
  const size_t a0 = (size_t)(t0 < 0 ? 0 : t0) * D_MODEL;
  const size_t a1 = (size_t)(t1 < 0 ? 0 : t1) * D_MODEL;

  // per-lane hbf row offsets for GLU stores (C layout: row = r + 8*hi (+16 for subtile 1))
  int rb0[8], rb1[8];
#pragma unroll
  for (int r = 0; r < 8; ++r) {
    int q0 = toks[r + 8 * hi];
    int q1 = toks[r + 8 * hi + 16];
    rb0[r] = q0 < 0 ? -1 : q0 * F_DIM;
    rb1[r] = q1 < 0 ? -1 : q1 * F_DIM;
  }

  float ps0[8] = {}, ps1[8] = {}, pq0[8] = {}, pq1[8] = {};

  for (int cc = 0; cc < 16; ++cc) {
    const int fa = cc * 128 + wave * 16 + m16;  // a-half column of this lane
    const int fb = fa + F_DIM;                  // matching gate column
    const __bf16* bA = w1t + ((size_t)e * TWO_F + fa) * D_MODEL;
    const __bf16* bB = w1t + ((size_t)e * TWO_F + fb) * D_MODEL;

    v8f zc = {};
    v8f acc_a0 = zc, acc_b0 = zc, acc_a1 = zc, acc_b1 = zc;
    for (int kk = 0; kk < D_MODEL; kk += 32) {
      __builtin_prefetch(bA + kk + 512, 0, 1);
      __builtin_prefetch(bB + kk + 512, 0, 1);
      Frag fA0, fA1, fBa, fBb;
      fA0.q[0] = *(const u32x4*)(xbf + a0 + kk + kb);
      fA0.q[1] = *(const u32x4*)(xbf + a0 + kk + kb + 16);
      fA1.q[0] = *(const u32x4*)(xbf + a1 + kk + kb);
      fA1.q[1] = *(const u32x4*)(xbf + a1 + kk + kb + 16);
      fBa.q[0] = *(const u32x4*)(bA + kk + kb);
      fBa.q[1] = *(const u32x4*)(bA + kk + kb + 16);
      fBb.q[0] = *(const u32x4*)(bB + kk + kb);
      fBb.q[1] = *(const u32x4*)(bB + kk + kb + 16);
      acc_a0 = wmma_bf16(fA0.v, fBa.v, acc_a0);
      acc_b0 = wmma_bf16(fA0.v, fBb.v, acc_b0);
      acc_a1 = wmma_bf16(fA1.v, fBa.v, acc_a1);
      acc_b1 = wmma_bf16(fA1.v, fBb.v, acc_b1);
    }
    const float biasA = b1[e * TWO_F + fa];
    const float biasB = b1[e * TWO_F + fb];
#pragma unroll
    for (int r = 0; r < 8; ++r) {
      float g0 = (acc_a0[r] + biasA) / (1.f + __expf(-(acc_b0[r] + biasB)));
      float g1 = (acc_a1[r] + biasA) / (1.f + __expf(-(acc_b1[r] + biasB)));
      ps0[r] += g0; pq0[r] += g0 * g0;
      ps1[r] += g1; pq1[r] += g1 * g1;
      if (rb0[r] >= 0) hbf[rb0[r] + fa] = (__bf16)g0;
      if (rb1[r] >= 0) hbf[rb1[r] + fa] = (__bf16)g1;
    }
  }

  // reduce per-row partial sums across the 16 lanes of each half-wave
#pragma unroll
  for (int off = 8; off >= 1; off >>= 1)
#pragma unroll
    for (int r = 0; r < 8; ++r) {
      ps0[r] += __shfl_xor(ps0[r], off, 32);
      pq0[r] += __shfl_xor(pq0[r], off, 32);
      ps1[r] += __shfl_xor(ps1[r], off, 32);
      pq1[r] += __shfl_xor(pq1[r], off, 32);
    }
  if (m16 == 0) {
#pragma unroll
    for (int r = 0; r < 8; ++r) {
      atomicAdd(&sums [r + 8 * hi],      ps0[r]);
      atomicAdd(&sumsq[r + 8 * hi],      pq0[r]);
      atomicAdd(&sums [r + 8 * hi + 16], ps1[r]);
      atomicAdd(&sumsq[r + 8 * hi + 16], pq1[r]);
    }
  }
  __syncthreads();
  if (tid < 32) {
    int tok = toks[tid];
    if (tok >= 0) {
      float mu  = sums[tid] / (float)F_DIM;
      float var = sumsq[tid] / (float)F_DIM - mu * mu;
      stats[2 * tok]     = mu;
      stats[2 * tok + 1] = rsqrtf(var + LN_EPS);
    }
  }
}

// ---------------- LayerNorm apply (in-place on hbf) ----------------
__global__ void k_norm(__bf16* __restrict__ hbf, const float* __restrict__ stats,
                       const int* __restrict__ choice,
                       const float* __restrict__ lng, const float* __restrict__ lnb) {
  int i = blockIdx.x * blockDim.x + threadIdx.x;
  int n = i >> 11;                 // F_DIM = 2048
  int c = i & (F_DIM - 1);
  int e = choice[n];
  float mu = stats[2 * n], rs = stats[2 * n + 1];
  float v = (float)hbf[i];
  hbf[i] = (__bf16)((v - mu) * rs * lng[e * F_DIM + c] + lnb[e * F_DIM + c]);
}

// ---------------- projection GEMM2: out = h @ Wp + bp (M=64 tile, bf16 WMMA) ----------------
__global__ void __launch_bounds__(256)
k_proj(const __bf16* __restrict__ hbf, const __bf16* __restrict__ wpt,
       const float* __restrict__ bp, float* __restrict__ out) {
  const int rt = blockIdx.x;           // 64-row tile
  const int tid = threadIdx.x, wave = tid >> 5, lane = tid & 31;
  const int m16 = lane & 15, hi = lane >> 4, kb = hi * 8;
  size_t arow[4];
#pragma unroll
  for (int ms = 0; ms < 4; ++ms)
    arow[ms] = ((size_t)rt * 64 + 16 * ms + m16) * F_DIM;

  for (int cc = 0; cc < 8; ++cc) {
    const int c0 = cc * 128 + wave * 16 + m16;
    const __bf16* bcol = wpt + (size_t)c0 * F_DIM;
    v8f zc = {};
    v8f acc[4] = {zc, zc, zc, zc};
    for (int kk = 0; kk < F_DIM; kk += 32) {
      __builtin_prefetch(bcol + kk + 512, 0, 1);
      Frag fB;
      fB.q[0] = *(const u32x4*)(bcol + kk + kb);
      fB.q[1] = *(const u32x4*)(bcol + kk + kb + 16);
#pragma unroll
      for (int ms = 0; ms < 4; ++ms) {   // one B fragment feeds 4 row-subtiles
        Frag fA;
        fA.q[0] = *(const u32x4*)(hbf + arow[ms] + kk + kb);
        fA.q[1] = *(const u32x4*)(hbf + arow[ms] + kk + kb + 16);
        acc[ms] = wmma_bf16(fA.v, fB.v, acc[ms]);
      }
    }
    const float bias = bp[c0];
#pragma unroll
    for (int ms = 0; ms < 4; ++ms)
#pragma unroll
      for (int r = 0; r < 8; ++r) {
        int m = 16 * ms + 8 * hi + r;
        out[((size_t)rt * 64 + m) * D_MODEL + c0] = acc[ms][r] + bias;
      }
  }
}

// ---------------- host side ----------------
extern "C" void kernel_launch(void* const* d_in, const int* in_sizes, int n_in,
                              void* d_out, int out_size, void* d_ws, size_t ws_size,
                              hipStream_t stream) {
  const float* x   = (const float*)d_in[0];
  const float* W1  = (const float*)d_in[1];
  const float* b1  = (const float*)d_in[2];
  const float* lng = (const float*)d_in[3];
  const float* lnb = (const float*)d_in[4];
  const float* Wg1 = (const float*)d_in[5];
  const float* bg1 = (const float*)d_in[6];
  const float* Wg2 = (const float*)d_in[7];
  const float* Wp  = (const float*)d_in[8];
  const float* bp  = (const float*)d_in[9];
  float* out = (float*)d_out;

  char* ws = (char*)d_ws;
  // workspace layout (~116.4 MiB total)
  __bf16* w1t   = (__bf16*)(ws);                                  // 67,108,864 B
  __bf16* wpt   = (__bf16*)(ws + 67108864);                       //  4,194,304 B
  __bf16* xbf   = (__bf16*)(ws + 71303168);                       // 16,777,216 B
  __bf16* hbf   = (__bf16*)(ws + 88080384);                       // 33,554,432 B
  int*    choice  = (int*)(ws + 121634816);                       //     32,768 B
  int*    sorted  = (int*)(ws + 121667584);                       //    262,144 B
  int*    counts  = (int*)(ws + 121929728);                       // 8 ints
  int*    cursors = counts + 8;                                   // 8 ints
  int*    padded  = counts + 16;                                  // 8 ints
  float*  stats   = (float*)(ws + 121929984);                     //     65,536 B

  // 1) one-time bf16 conversions (transposed weights)
  k_tw1<<<dim3(TWO_F / 32, D_MODEL / 32, NUM_E), 256, 0, stream>>>(W1, w1t);
  k_twp<<<dim3(D_MODEL / 32, F_DIM / 32), 256, 0, stream>>>(Wp, wpt);
  k_cx<<<(N_TOK * D_MODEL) / 256, 256, 0, stream>>>(x, xbf, N_TOK * D_MODEL);

  // 2) routing (exact f32) + balance loss + bucket sort by expert
  k_zero<<<1, 32, 0, stream>>>(counts);
  k_gate<<<N_TOK / 8, 256, 0, stream>>>(x, Wg1, bg1, Wg2, choice, counts);
  k_loss<<<1, 32, 0, stream>>>(counts, out + (size_t)N_TOK * D_MODEL);
  k_sort<<<N_TOK / 256, 256, 0, stream>>>(choice, cursors, sorted);
  k_pad<<<1, 32, 0, stream>>>(counts, padded, sorted);

  // 3) routed expert FFN (GEMM1 + GLU + LN stats), then LN apply
  k_ffn<<<dim3(N_TOK / 32, NUM_E), 256, 0, stream>>>(xbf, w1t, b1, sorted, padded,
                                                     hbf, stats);
  k_norm<<<(N_TOK * F_DIM) / 256, 256, 0, stream>>>(hbf, stats, choice, lng, lnb);

  // 4) projection (64-row tiles: one B fragment feeds 4 row-subtile WMMAs)
  k_proj<<<N_TOK / 64, 256, 0, stream>>>(hbf, wpt, bp, out);
}
